// CrossAttention_31920196944220
// MI455X (gfx1250) — matrix operand order repaired and checked by reading, not compiled, not run on hardware
//
#include <hip/hip_runtime.h>

// ---------------- static problem dimensions ----------------
#define HWn     2560
#define Tn      16
#define HEADSn  8
#define DHn     64
#define QDn     512
#define MROWS   40960          // HW*T
#define NBATCH  20480          // HW*HEADS
#define QKVSZ   20971520       // NBATCH*T*DH elements (bf16)

typedef __attribute__((ext_vector_type(16))) __bf16 bf16x16;
typedef __attribute__((ext_vector_type(8)))  float  f32x8;

union Frag {
  bf16x16 v;
  uint4   u[2];
  unsigned short h[16];
};

__device__ __forceinline__ unsigned short f2bf(float f) {
  union { float f; unsigned int u; } c; c.f = f;
  unsigned int u = c.u;
  u += 0x7fffu + ((u >> 16) & 1u);           // round-to-nearest-even
  return (unsigned short)(u >> 16);
}
__device__ __forceinline__ unsigned int packbf2(float a, float b) {
  return (unsigned int)f2bf(a) | ((unsigned int)f2bf(b) << 16);
}
__device__ __forceinline__ f32x8 wmma_bf16(const Frag& a, const Frag& b, f32x8 c) {
  return __builtin_amdgcn_wmma_f32_16x16x32_bf16(false, a.v, false, b.v,
                                                 (short)0, c, false, false);
}

// gfx1250 async global->LDS copy (16B per lane), tracked by ASYNCcnt.
// lds_off = LDS byte offset (low 32 bits of a generic pointer to __shared__).
__device__ __forceinline__ void async_copy16(unsigned lds_off, const void* gptr) {
  asm volatile("global_load_async_to_lds_b128 %0, %1, off"
               :: "v"(lds_off), "v"(gptr) : "memory");
}
__device__ __forceinline__ void wait_async0() {
  asm volatile("s_wait_asynccnt 0x0" ::: "memory");
}
__device__ __forceinline__ unsigned lds_off_of(const void* p) {
  return (unsigned)(uintptr_t)p;   // generic LDS ptr = {aperture, 32-bit offset}
}

// ---------------- fp32 -> bf16 pack (8 elems / thread) ----------------
__global__ __launch_bounds__(256) void cvt_f32_to_bf16(
    const float* __restrict__ src, unsigned short* __restrict__ dst, int n8) {
  int i = blockIdx.x * 256 + threadIdx.x;
  if (i >= n8) return;
  const float4* s = (const float4*)src + (size_t)i * 2;
  float4 a = s[0], b = s[1];
  uint4 o;
  o.x = packbf2(a.x, a.y); o.y = packbf2(a.z, a.w);
  o.z = packbf2(b.x, b.y); o.w = packbf2(b.z, b.w);
  ((uint4*)dst)[i] = o;
}

// ---------------- bf16 WMMA GEMM: C[M,N] = A[M,512] * B[512,N] ----------------
// A tile staged via async global->LDS (byte-identical row-major copy).
// B tile staged through regs (transposed into LDS for conflict-free frag reads).
// MODE 0: fused QKV (N spans 3 weights of 512); scatter to head-split bf16
//         (Q,K row-major [b][t][64]; V transposed [b][d][16]).
// MODE 1: out-projection; f32 output + bias, row-major.
template<int MODE>
__global__ __launch_bounds__(256) void gemm_bf16_wmma(
    const unsigned short* __restrict__ A,     // [MROWS][512] bf16
    const unsigned short* __restrict__ Bmat,  // [nW][512][512] bf16
    unsigned short* __restrict__ qkv,         // MODE 0 output base (Q|K|V contiguous)
    float* __restrict__ Cout,                 // MODE 1 output
    const float* __restrict__ bias)           // MODE 1 bias
{
  __shared__ unsigned short Alds[2][128 * 40];   // 128 rows, stride 40 (pad)
  __shared__ unsigned short Blds[2][64 * 40];    // [n][k], stride 40 (pad)

  const int tid  = threadIdx.x;
  const int lane = tid & 31, wave = tid >> 5;
  const int wm = wave >> 1, wn = wave & 1;       // 4x2 wave grid
  const int m0   = blockIdx.x * 128;
  const int n0g  = blockIdx.y * 64;
  const int wsel = n0g >> 9;                     // which weight (MODE 0)
  const int ncol0 = n0g & 511;
  const unsigned short* Bbase = Bmat + wsel * (512 * 512) + ncol0;

  const int ra = tid >> 1, ca = (tid & 1) * 16;  // A staging: 32B/thread
  const int kbr = tid >> 3, cb = (tid & 7) * 8;  // B staging: 16B/thread
  const unsigned short* Aptr = A + (m0 + ra) * 512 + ca;

  unsigned ldsA[2];
  ldsA[0] = lds_off_of(&Alds[0][ra * 40 + ca]);
  ldsA[1] = lds_off_of(&Alds[1][ra * 40 + ca]);

  // prologue: async A tile 0 into buf 0; B tile 0 into regs
  async_copy16(ldsA[0],      Aptr);
  async_copy16(ldsA[0] + 16, Aptr + 8);
  uint4 sb0 = *(const uint4*)(Bbase + kbr * 512 + cb);

  const f32x8 zf = {0.f,0.f,0.f,0.f,0.f,0.f,0.f,0.f};
  f32x8 acc[2][2] = {{zf, zf}, {zf, zf}};

  const int rsel = lane & 15;
  const int h8 = (lane >> 4) * 8, h16 = (lane >> 4) * 16;

  int buf = 0;
  for (int kk = 0; kk < 512; kk += 32) {
    // stage current B tile to LDS (transposed)
    {
      const unsigned short* pb = (const unsigned short*)&sb0;
      #pragma unroll
      for (int j = 0; j < 8; ++j) Blds[buf][(cb + j) * 40 + kbr] = pb[j];
    }
    wait_async0();             // A tile kk resident in Alds[buf]
    __syncthreads();
    if (kk + 32 < 512) {       // async A tile kk+32 into the other buffer
      async_copy16(ldsA[buf ^ 1],      Aptr + kk + 32);
      async_copy16(ldsA[buf ^ 1] + 16, Aptr + kk + 40);
      sb0 = *(const uint4*)(Bbase + (kk + 32 + kbr) * 512 + cb);
    }
    if (kk + 64 < 512) {       // gfx1250 global_prefetch_b8, two tiles ahead
      __builtin_prefetch(Aptr + (kk + 64), 0, 1);
      __builtin_prefetch(Bbase + (kk + 64 + kbr) * 512 + cb, 0, 1);
    }
    // fragments + 2x2 WMMA
    Frag af[2], bfr[2];
    #pragma unroll
    for (int mt = 0; mt < 2; ++mt) {
      const unsigned short* p = &Alds[buf][(wm * 32 + mt * 16 + rsel) * 40 + h8];
      af[mt].u[0] = *(const uint4*)(p);
      af[mt].u[1] = *(const uint4*)(p + 16);
    }
    #pragma unroll
    for (int nt = 0; nt < 2; ++nt) {
      const unsigned short* p = &Blds[buf][(wn * 32 + nt * 16 + rsel) * 40 + h8];
      bfr[nt].u[0] = *(const uint4*)(p);
      bfr[nt].u[1] = *(const uint4*)(p + 16);
    }
    #pragma unroll
    for (int mt = 0; mt < 2; ++mt)
      #pragma unroll
      for (int nt = 0; nt < 2; ++nt)
        acc[mt][nt] = wmma_bf16(af[mt], bfr[nt], acc[mt][nt]);
    buf ^= 1;
  }

  // ---- store ----
  const int h4 = lane >> 4;
  #pragma unroll
  for (int mt = 0; mt < 2; ++mt) {
    #pragma unroll
    for (int nt = 0; nt < 2; ++nt) {
      #pragma unroll
      for (int r = 0; r < 8; ++r) {
        int m = m0 + wm * 32 + mt * 16 + r + 8 * h4;
        int nloc = wn * 32 + nt * 16 + rsel;
        float val = acc[mt][nt][r];
        if (MODE == 0) {
          int nn = ncol0 + nloc;
          int head = nn >> 6, d = nn & 63;
          int hw = m >> 4, t = m & 15;
          int b = hw * 8 + head;
          if (wsel < 2)   // Q, K: [b][t][64]
            qkv[wsel * QKVSZ + (b * 16 + t) * 64 + d] = f2bf(val);
          else            // V transposed: [b][d][16]
            qkv[2 * QKVSZ + (b * 64 + d) * 16 + t] = f2bf(val);
        } else {
          int n = n0g + nloc;
          Cout[m * 512 + n] = val + bias[n];
        }
      }
    }
  }
}

// ---------------- fused attention: one wave per (hw,head) batch ----------------
__global__ __launch_bounds__(256) void attn_wmma(
    const unsigned short* __restrict__ Qs,   // [b][t][64] bf16
    const unsigned short* __restrict__ Ks,   // [b][t][64] bf16
    const unsigned short* __restrict__ Vt,   // [b][d][16] bf16
    const float* __restrict__ rel_k,         // [33][64] f32 (rows 0..31 used)
    const float* __restrict__ rel_v,
    unsigned short* __restrict__ AttOut)     // [hw][t][512] bf16
{
  __shared__ unsigned short relkL[32 * 72];   // row-major, padded stride
  __shared__ unsigned short relvTL[64 * 40];  // transposed [d][r], padded stride
  __shared__ float attnL[8][16 * 17];         // per-wave attn, padded stride

  const int tid = threadIdx.x;
  for (int i = tid; i < 2048; i += 256) {
    int r = i >> 6, d = i & 63;
    relkL[r * 72 + d]  = f2bf(rel_k[r * 64 + d]);
    relvTL[d * 40 + r] = f2bf(rel_v[r * 64 + d]);
  }
  __syncthreads();

  const int wave = tid >> 5, lane = tid & 31;
  const int b = blockIdx.x * 8 + wave;
  const int hw = b >> 3, head = b & 7;
  const int y = hw / 40, xp = hw % 40;

  // trajectory fg/bg bitmask over 16 frames (box lerp .10->.60, 16x10 box)
  unsigned fgbits = 0;
  #pragma unroll
  for (int t = 0; t < 16; ++t) {
    float r = (float)t * (1.0f / 15.0f);
    int hs  = (int)((0.10f + 0.5f * r) * 64.0f);
    int wss = (int)((0.10f + 0.5f * r) * 40.0f);
    if (y >= hs && y < hs + 16 && xp >= wss && xp < wss + 10) fgbits |= 1u << t;
  }

  const int tA = lane & 15, h4 = lane >> 4;
  const int c0 = h4 * 8, k0 = h4 * 16;

  // q A-fragments (K 0..31 and 32..63)
  Frag qa0, qa1;
  const unsigned short* qrow = Qs + b * 1024 + tA * 64;
  qa0.u[0] = *(const uint4*)(qrow + k0);
  qa0.u[1] = *(const uint4*)(qrow + k0 + 8);
  qa1.u[0] = *(const uint4*)(qrow + 32 + k0);
  qa1.u[1] = *(const uint4*)(qrow + 32 + k0 + 8);

  // k^T B-fragments
  Frag kb0, kb1;
  const unsigned short* krow = Ks + b * 1024 + tA * 64;
  kb0.u[0] = *(const uint4*)(krow + k0);
  kb0.u[1] = *(const uint4*)(krow + k0 + 8);
  kb1.u[0] = *(const uint4*)(krow + 32 + k0);
  kb1.u[1] = *(const uint4*)(krow + 32 + k0 + 8);

  const f32x8 zf = {0.f,0.f,0.f,0.f,0.f,0.f,0.f,0.f};
  f32x8 sim = zf;
  sim = wmma_bf16(qa0, kb0, sim);
  sim = wmma_bf16(qa1, kb1, sim);

  // QR = q * rel_k^T  (cols 0..15 and 16..31)
  Frag rlo, rhi;
  const unsigned short* rk0 = relkL + tA * 72;
  rlo.u[0] = *(const uint4*)(rk0 + k0);      rlo.u[1] = *(const uint4*)(rk0 + k0 + 8);
  rhi.u[0] = *(const uint4*)(rk0 + 32 + k0); rhi.u[1] = *(const uint4*)(rk0 + 32 + k0 + 8);
  f32x8 QR0 = zf;
  QR0 = wmma_bf16(qa0, rlo, QR0);
  QR0 = wmma_bf16(qa1, rhi, QR0);
  const unsigned short* rk1 = relkL + (16 + tA) * 72;
  rlo.u[0] = *(const uint4*)(rk1 + k0);      rlo.u[1] = *(const uint4*)(rk1 + k0 + 8);
  rhi.u[0] = *(const uint4*)(rk1 + 32 + k0); rhi.u[1] = *(const uint4*)(rk1 + 32 + k0 + 8);
  f32x8 QR1 = zf;
  QR1 = wmma_bf16(qa0, rlo, QR1);
  QR1 = wmma_bf16(qa1, rhi, QR1);

  // bias gather (lane permute), mask, scale, softmax
  float attnv[8];
  #pragma unroll
  for (int r = 0; r < 8; ++r) {
    int t = r + 8 * h4;
    int s = tA;
    int c = s - t + 16;                       // in [1,31]
    int srcLane = (lane & 16) | (c & 15);
    float g0 = __shfl(QR0[r], srcLane, 32);
    float g1 = __shfl(QR1[r], srcLane, 32);
    float rel = (c >= 16) ? g1 : g0;
    float mk = (((fgbits >> t) ^ (fgbits >> s)) & 1u) ? 0.01f : 1.0f;
    attnv[r] = (sim[r] + rel) * 0.125f * mk;
  }
  #pragma unroll
  for (int r = 0; r < 8; ++r) {
    float mx = attnv[r];
    #pragma unroll
    for (int o = 8; o > 0; o >>= 1) mx = fmaxf(mx, __shfl_xor(mx, o, 16));
    float p = __expf(attnv[r] - mx);
    float sm = p;
    #pragma unroll
    for (int o = 8; o > 0; o >>= 1) sm += __shfl_xor(sm, o, 16);
    attnv[r] = p / sm;
  }

  // stage attn to LDS for C->A layout transpose
  float* aL = attnL[wave];
  #pragma unroll
  for (int r = 0; r < 8; ++r) aL[(r + 8 * h4) * 17 + tA] = attnv[r];
  __syncthreads();

  // attn A-frag (K 0..15 live, 16..31 zero) and shifted AS A-frag (K=32)
  Frag fa, fas;
  {
    const float* row = aL + tA * 17;
    fa.u[0].x = packbf2(row[c0 + 0], row[c0 + 1]);
    fa.u[0].y = packbf2(row[c0 + 2], row[c0 + 3]);
    fa.u[0].z = packbf2(row[c0 + 4], row[c0 + 5]);
    fa.u[0].w = packbf2(row[c0 + 6], row[c0 + 7]);
    fa.u[1] = make_uint4(0u, 0u, 0u, 0u);
    float tmp[16];
    #pragma unroll
    for (int j = 0; j < 8; ++j) {
      int i1 = (c0 + j) + tA - 16;            // AS[t, c0+j]
      int i2 = (c0 + 16 + j) + tA - 16;       // AS[t, c0+16+j]
      tmp[j]     = (i1 >= 0 && i1 < 16) ? row[i1] : 0.0f;
      tmp[8 + j] = (i2 >= 0 && i2 < 16) ? row[i2] : 0.0f;
    }
    fas.u[0].x = packbf2(tmp[0],  tmp[1]);
    fas.u[0].y = packbf2(tmp[2],  tmp[3]);
    fas.u[0].z = packbf2(tmp[4],  tmp[5]);
    fas.u[0].w = packbf2(tmp[6],  tmp[7]);
    fas.u[1].x = packbf2(tmp[8],  tmp[9]);
    fas.u[1].y = packbf2(tmp[10], tmp[11]);
    fas.u[1].z = packbf2(tmp[12], tmp[13]);
    fas.u[1].w = packbf2(tmp[14], tmp[15]);
  }

  // out = attn*v + AS*rel_v, 4 N-tiles over d
  #pragma unroll
  for (int nt = 0; nt < 4; ++nt) {
    Frag bv, brv;
    {
      const unsigned short* vr = Vt + b * 1024 + (nt * 16 + tA) * 16;
      bv.u[0] = *(const uint4*)(vr + c0);
      bv.u[1] = make_uint4(0u, 0u, 0u, 0u);
    }
    const unsigned short* rv = relvTL + (nt * 16 + tA) * 40 + c0;
    brv.u[0] = *(const uint4*)(rv);
    brv.u[1] = *(const uint4*)(rv + 16);
    f32x8 o = zf;
    o = wmma_bf16(fa, bv, o);
    o = wmma_bf16(fas, brv, o);
    #pragma unroll
    for (int r = 0; r < 8; ++r) {
      int t = r + 8 * h4;
      int col = head * 64 + nt * 16 + tA;
      AttOut[(hw * 16 + t) * 512 + col] = f2bf(o[r]);
    }
  }
}

// ---------------- launcher ----------------
extern "C" void kernel_launch(void* const* d_in, const int* in_sizes, int n_in,
                              void* d_out, int out_size, void* d_ws, size_t ws_size,
                              hipStream_t stream) {
  const float* x     = (const float*)d_in[0];
  const float* Wq    = (const float*)d_in[1];
  const float* Wk    = (const float*)d_in[2];
  const float* Wv    = (const float*)d_in[3];
  const float* Wo    = (const float*)d_in[4];
  const float* bo    = (const float*)d_in[5];
  const float* rel_k = (const float*)d_in[6];
  const float* rel_v = (const float*)d_in[7];

  unsigned short* ws   = (unsigned short*)d_ws;
  unsigned short* Xb   = ws;                              // MROWS*512
  unsigned short* Wb   = Xb + (size_t)MROWS * 512;        // 4*512*512
  unsigned short* Qs   = Wb + 4 * 512 * 512;              // 3*QKVSZ (Q|K|V)
  unsigned short* AttO = Qs + (size_t)3 * QKVSZ;          // MROWS*512

  // fp32 -> bf16 packing
  cvt_f32_to_bf16<<<(MROWS * 512 / 8 + 255) / 256, 256, 0, stream>>>(x, Xb, MROWS * 512 / 8);
  cvt_f32_to_bf16<<<128, 256, 0, stream>>>(Wq, Wb + 0 * 262144, 32768);
  cvt_f32_to_bf16<<<128, 256, 0, stream>>>(Wk, Wb + 1 * 262144, 32768);
  cvt_f32_to_bf16<<<128, 256, 0, stream>>>(Wv, Wb + 2 * 262144, 32768);
  cvt_f32_to_bf16<<<128, 256, 0, stream>>>(Wo, Wb + 3 * 262144, 32768);

  // fused QKV projection (N = 3*512)
  gemm_bf16_wmma<0><<<dim3(MROWS / 128, 24), 256, 0, stream>>>(
      Xb, Wb, Qs, nullptr, nullptr);

  // fused attention (one wave per batch, 8 waves/block)
  attn_wmma<<<NBATCH / 8, 256, 0, stream>>>(
      Qs, Qs + QKVSZ, Qs + 2 * QKVSZ, rel_k, rel_v, AttO);

  // output projection + bias -> f32
  gemm_bf16_wmma<1><<<dim3(MROWS / 128, 8), 256, 0, stream>>>(
      AttO, Wb + 3 * 262144, nullptr, (float*)d_out, bo);
}
